// CICDM_Net_80135499809345
// MI455X (gfx1250) — matrix operand order repaired and checked by reading, not compiled, not run on hardware
//
#include <hip/hip_runtime.h>

// CICDM on MI455X (gfx1250): bf16 WMMA pipeline, f32 accumulate.
// B=1024 L=256 E=8192 C=512 P=32
#define NB 1024
#define NL 256
#define NE 8192
#define NC 512
#define NP 32

typedef __attribute__((ext_vector_type(16))) __bf16       v16bf;
typedef __attribute__((ext_vector_type(8)))  float        v8f;
typedef __attribute__((ext_vector_type(4)))  unsigned int v4u;

union Frag32B { v4u u[2]; v16bf bf; };

__device__ __forceinline__ float sigm(float x) { return 1.0f / (1.0f + __expf(-x)); }

__device__ __forceinline__ unsigned short f2bf(float f) {
  unsigned int u = __float_as_uint(f);
  u += 0x7fffu + ((u >> 16) & 1u);           // round-to-nearest-even
  return (unsigned short)(u >> 16);
}

// A-operand fragment, 16x32 bf16 (ISA 7.12.2): lane m=lane&15, halves split K in 8s.
__device__ __forceinline__ v16bf load_frag_a16(const unsigned short* base, int ld,
                                               int row0, int k0, int lane) {
  int m = lane & 15, h = lane >> 4;
  const unsigned short* p = base + (size_t)(row0 + m) * ld + k0 + h * 8;
  Frag32B f;
  f.u[0] = *(const v4u*)p;          // K = k0+h*8 .. +7      -> VGPR 0..3
  f.u[1] = *(const v4u*)(p + 16);   // K = k0+16+h*8 .. +7   -> VGPR 4..7
  return f.bf;
}

// B-operand fragment, 32x16 bf16: lane n=lane&15 owns column n; halves split K in 8s
// (same as A). Source stored "by column": Bsrc[n][k] row-major, stride ld.
__device__ __forceinline__ v16bf load_frag_b16(const unsigned short* base, int ld,
                                               int n0, int k0, int lane) {
  int n = lane & 15, h = lane >> 4;
  const unsigned short* p = base + (size_t)(n0 + n) * ld + k0 + h * 8;
  Frag32B f;
  f.u[0] = *(const v4u*)p;
  f.u[1] = *(const v4u*)(p + 16);
  return f.bf;
}

__device__ __forceinline__ v8f wmma_bf16(v16bf a, v16bf b, v8f c) {
  return __builtin_amdgcn_wmma_f32_16x16x32_bf16(false, a, false, b, (short)0, c,
                                                 false, false);
}

// ---------------------------------------------------------------------------
// K1: W = sigmoid(exer_conc_w)*adj; W2 = W/rowsum(W)  (W f32, W2 bf16, both [E,C])
__global__ void prep_w(const float* __restrict__ ecw, const float* __restrict__ adj,
                       float* __restrict__ W, unsigned short* __restrict__ W2bf) {
  int e = blockIdx.x, t = threadIdx.x;
  size_t ro = (size_t)e * NC;
  float w0 = sigm(ecw[ro + t])        * adj[ro + t];
  float w1 = sigm(ecw[ro + t + 256])  * adj[ro + t + 256];
  float s = w0 + w1;
  #pragma unroll
  for (int m = 16; m > 0; m >>= 1) s += __shfl_xor(s, m, 32);
  __shared__ float ls[8];
  if ((t & 31) == 0) ls[t >> 5] = s;
  __syncthreads();
  float tot = 0.f;
  #pragma unroll
  for (int i = 0; i < 8; i++) tot += ls[i];
  float inv = 1.0f / tot;   // >=1 concept per exercise and sigmoid>0 -> tot>0
  W[ro + t] = w0;           W[ro + t + 256] = w1;
  W2bf[ro + t] = f2bf(w0 * inv); W2bf[ro + t + 256] = f2bf(w1 * inv);
}

// K2: D2 = softmax(exer_pote_w, axis=P) -> bf16 [E,P]. One wave32 per exercise.
__global__ void d2_softmax(const float* __restrict__ pote,
                           unsigned short* __restrict__ D2bf) {
  int lane = threadIdx.x & 31;
  int e = blockIdx.x * 8 + (threadIdx.x >> 5);
  float v = pote[(size_t)e * NP + lane];
  float mx = v;
  #pragma unroll
  for (int m = 16; m > 0; m >>= 1) mx = fmaxf(mx, __shfl_xor(mx, m, 32));
  float ex = __expf(v - mx);
  float sm = ex;
  #pragma unroll
  for (int m = 16; m > 0; m >>= 1) sm += __shfl_xor(sm, m, 32);
  D2bf[(size_t)e * NP + lane] = f2bf(ex / sm);
}

// K3a: per-column max of conc_conc_w [C,C]
__global__ void colmax_k(const float* __restrict__ cc, float* __restrict__ cmax) {
  int d = blockIdx.x * 256 + threadIdx.x;
  float m = -3.4e38f;
  for (int r = 0; r < NC; r++) m = fmaxf(m, cc[(size_t)r * NC + d]);
  cmax[d] = m;
}

// K3b: eT[n,k] = exp(cc[k,n] - colmax[n])  (bf16, transposed for B-fragment loads)
__global__ void build_eT(const float* __restrict__ cc, const float* __restrict__ cmax,
                         unsigned short* __restrict__ eT) {
  int k = blockIdx.x, t = threadIdx.x;
  #pragma unroll
  for (int i = 0; i < 2; i++) {
    int n = t + i * 256;
    float v = __expf(cc[(size_t)k * NC + n] - cmax[n]);
    eT[(size_t)n * NC + k] = f2bf(v);
  }
}

// K4: gather-reduce: a[b,c] = (sum_l score*W[e_l,c]) / (sum_l W[e_l,c]), mask flag.
__global__ void gather_a(const int* __restrict__ exer, const float* __restrict__ score,
                         const float* __restrict__ W,
                         unsigned short* __restrict__ a_bf,
                         unsigned short* __restrict__ m_bf) {
  int b = blockIdx.x, t = threadIdx.x;
  __shared__ int   se[NL];
  __shared__ float ss[NL];
  se[t] = exer[(size_t)b * NL + t];
  ss[t] = score[(size_t)b * NL + t];
  __syncthreads();
  float s0 = 0.f, s1 = 0.f, n0 = 0.f, n1 = 0.f;
  for (int l = 0; l < NL; l++) {
    const float* row = W + (size_t)se[l] * NC;
    if (l + 1 < NL)
      __builtin_prefetch((const void*)(W + (size_t)se[l + 1] * NC + t), 0, 0);
    float w0 = row[t], w1 = row[t + 256], sc = ss[l];
    s0 += w0; s1 += w1; n0 += sc * w0; n1 += sc * w1;
  }
  size_t o = (size_t)b * NC + t;
  a_bf[o]       = f2bf(s0 > 0.f ? n0 / s0 : 0.f);
  m_bf[o]       = f2bf(s0 > 0.f ? 1.f : 0.f);
  a_bf[o + 256] = f2bf(s1 > 0.f ? n1 / s1 : 0.f);
  m_bf[o + 256] = f2bf(s1 > 0.f ? 1.f : 0.f);
}

// K5: Bm[b,p] = sum_l score * softmax_P(pote[e_l])[p]  (wave-parallel softmax)
__global__ void bm_kernel(const int* __restrict__ exer, const float* __restrict__ score,
                          const float* __restrict__ pote,
                          unsigned short* __restrict__ Bm_bf) {
  int b = blockIdx.x, wave = threadIdx.x >> 5, lane = threadIdx.x & 31;
  float acc = 0.f;
  for (int l = wave; l < NL; l += 8) {
    int   e  = exer[(size_t)b * NL + l];
    float sc = score[(size_t)b * NL + l];
    float v  = pote[(size_t)e * NP + lane];
    float mx = v;
    #pragma unroll
    for (int m = 16; m > 0; m >>= 1) mx = fmaxf(mx, __shfl_xor(mx, m, 32));
    float ex = __expf(v - mx);
    float sm = ex;
    #pragma unroll
    for (int m = 16; m > 0; m >>= 1) sm += __shfl_xor(sm, m, 32);
    acc += sc * ex / sm;
  }
  __shared__ float red[256];
  red[threadIdx.x] = acc;
  __syncthreads();
  if (wave == 0) {
    float tot = 0.f;
    #pragma unroll
    for (int w = 0; w < 8; w++) tot += red[w * 32 + lane];
    Bm_bf[(size_t)b * NP + lane] = f2bf(tot);
  }
}

// K6: A = (a@e)/(mask@e) via shared-B double WMMA GEMM. 8 waves x (16x32) tiles.
__global__ void a_gemm(const unsigned short* __restrict__ a_bf,
                       const unsigned short* __restrict__ m_bf,
                       const unsigned short* __restrict__ eT,
                       float* __restrict__ Aout, unsigned short* __restrict__ Abf) {
  int lane = threadIdx.x & 31, wave = threadIdx.x >> 5;
  int m0 = blockIdx.y * 16;
  int nb = blockIdx.x * 256 + wave * 32;
  v8f aN0 = {}, aN1 = {}, aD0 = {}, aD1 = {};
  for (int k0 = 0; k0 < NC; k0 += 32) {
    v16bf af = load_frag_a16(a_bf, NC, m0, k0, lane);
    v16bf mf = load_frag_a16(m_bf, NC, m0, k0, lane);
    v16bf b0 = load_frag_b16(eT, NC, nb,      k0, lane);
    v16bf b1 = load_frag_b16(eT, NC, nb + 16, k0, lane);
    aN0 = wmma_bf16(af, b0, aN0); aN1 = wmma_bf16(af, b1, aN1);
    aD0 = wmma_bf16(mf, b0, aD0); aD1 = wmma_bf16(mf, b1, aD1);
  }
  int n = lane & 15, h = lane >> 4;
  #pragma unroll
  for (int r = 0; r < 8; r++) {
    int row = m0 + h * 8 + r;
    float v0 = (aD0[r] != 0.f) ? aN0[r] / aD0[r] : 0.f;
    float v1 = (aD1[r] != 0.f) ? aN1[r] / aD1[r] : 0.f;
    size_t o0 = (size_t)row * NC + nb + n, o1 = o0 + 16;
    Aout[o0] = v0; Abf[o0] = f2bf(v0);
    Aout[o1] = v1; Abf[o1] = f2bf(v1);
  }
}

// K7: Y = combine( A@W2^T (K=512), Bm@D2^T (K=32) ) + lambda/guess/slide epilogue.
// Register tiling: wave computes 32x64 (2 M-frags x 4 N-frags, 8 accumulators).
// Each B fragment feeds 2 WMMAs; block covers 32 M-rows so W2 is streamed from L2
// only (B/32)=32 times (~256 MB total) instead of 64.
__global__ void y_gemm(const unsigned short* __restrict__ Abf,
                       const unsigned short* __restrict__ W2bf,
                       const unsigned short* __restrict__ Bmbf,
                       const unsigned short* __restrict__ D2bf,
                       const float* __restrict__ lambd, const float* __restrict__ guess,
                       const float* __restrict__ slide, float* __restrict__ Y) {
  int lane = threadIdx.x & 31, wave = threadIdx.x >> 5;
  int m0 = blockIdx.y * 32;               // 2 M-fragments: m0, m0+16
  int nb = blockIdx.x * 512 + wave * 64;  // 4 N-fragments
  v8f accA[2][4] = {{{}, {}, {}, {}}, {{}, {}, {}, {}}};
  for (int k0 = 0; k0 < NC; k0 += 32) {
    v16bf a0 = load_frag_a16(Abf, NC, m0,      k0, lane);
    v16bf a1 = load_frag_a16(Abf, NC, m0 + 16, k0, lane);
    #pragma unroll
    for (int t = 0; t < 4; t++) {
      v16bf bf = load_frag_b16(W2bf, NC, nb + t * 16, k0, lane);
      accA[0][t] = wmma_bf16(a0, bf, accA[0][t]);
      accA[1][t] = wmma_bf16(a1, bf, accA[1][t]);
    }
  }
  v8f accB[2][4] = {{{}, {}, {}, {}}, {{}, {}, {}, {}}};
  {
    v16bf bm0 = load_frag_a16(Bmbf, NP, m0,      0, lane);
    v16bf bm1 = load_frag_a16(Bmbf, NP, m0 + 16, 0, lane);
    #pragma unroll
    for (int t = 0; t < 4; t++) {
      v16bf d2f = load_frag_b16(D2bf, NP, nb + t * 16, 0, lane);
      accB[0][t] = wmma_bf16(bm0, d2f, accB[0][t]);
      accB[1][t] = wmma_bf16(bm1, d2f, accB[1][t]);
    }
  }
  int n = lane & 15, h = lane >> 4;
  #pragma unroll
  for (int t = 0; t < 4; t++) {
    int e = nb + t * 16 + n;
    float lam = sigm(lambd[e]), sg = sigm(guess[e]), sl = sigm(slide[e]);
    #pragma unroll
    for (int mi = 0; mi < 2; mi++) {
      #pragma unroll
      for (int r = 0; r < 8; r++) {
        float y = (1.0f - lam) * accA[mi][t][r] + lam * accB[mi][t][r];
        y = fminf(fmaxf(y, 1e-8f), 1.0f - 1e-8f);
        y = (1.0f - sl) * y + sg * (1.0f - y);
        Y[(size_t)(m0 + mi * 16 + h * 8 + r) * NE + e] = y;
      }
    }
  }
}

// ---------------------------------------------------------------------------
extern "C" void kernel_launch(void* const* d_in, const int* in_sizes, int n_in,
                              void* d_out, int out_size, void* d_ws, size_t ws_size,
                              hipStream_t stream) {
  const int*   exer  = (const int*)d_in[0];
  const float* score = (const float*)d_in[1];
  // d_in[2] school_feature, d_in[3] school_feature_dim_w: unused downstream (per reference)
  const float* adj   = (const float*)d_in[4];
  const float* ecw   = (const float*)d_in[5];
  const float* cc    = (const float*)d_in[6];
  const float* pote  = (const float*)d_in[7];
  const float* lambd = (const float*)d_in[8];
  const float* guess = (const float*)d_in[9];
  const float* slide = (const float*)d_in[10];

  float* A_out = (float*)d_out;                       // [B, C]
  float* Y_out = (float*)d_out + (size_t)NB * NC;     // [B, E]

  // Workspace carve (~30 MB, 256B-aligned regions)
  char* base = (char*)d_ws;
  auto carve = [&](size_t bytes) {
    void* p = (void*)base;
    base += (bytes + 255) & ~(size_t)255;
    return p;
  };
  float*          W     = (float*)         carve((size_t)NE * NC * 4);  // 16 MB
  unsigned short* W2bf  = (unsigned short*)carve((size_t)NE * NC * 2);  //  8 MB
  unsigned short* D2bf  = (unsigned short*)carve((size_t)NE * NP * 2);
  unsigned short* a_bf  = (unsigned short*)carve((size_t)NB * NC * 2);
  unsigned short* m_bf  = (unsigned short*)carve((size_t)NB * NC * 2);
  unsigned short* Bmbf  = (unsigned short*)carve((size_t)NB * NP * 2);
  float*          cmax  = (float*)         carve((size_t)NC * 4);
  unsigned short* eT    = (unsigned short*)carve((size_t)NC * NC * 2);
  unsigned short* Abf   = (unsigned short*)carve((size_t)NB * NC * 2);

  prep_w    <<<NE,            256, 0, stream>>>(ecw, adj, W, W2bf);
  d2_softmax<<<NE / 8,        256, 0, stream>>>(pote, D2bf);
  colmax_k  <<<NC / 256,      256, 0, stream>>>(cc, cmax);
  build_eT  <<<NC,            256, 0, stream>>>(cc, cmax, eT);
  gather_a  <<<NB,            256, 0, stream>>>(exer, score, W, a_bf, m_bf);
  bm_kernel <<<NB,            256, 0, stream>>>(exer, score, pote, Bmbf);
  a_gemm    <<<dim3(NC / 256, NB / 16), 256, 0, stream>>>(a_bf, m_bf, eT, A_out, Abf);
  y_gemm    <<<dim3(NE / 512, NB / 32), 256, 0, stream>>>(Abf, W2bf, Bmbf, D2bf,
                                                          lambd, guess, slide, Y_out);
}
